// TransformerEncoderLayer_13030930776137
// MI455X (gfx1250) — matrix operand order repaired and checked by reading, not compiled, not run on hardware
//
#include <hip/hip_runtime.h>

// Transformer encoder layer, bf16 WMMA + async-LDS pipeline for gfx1250.
// B=4 S=2048 D=1024 H=16 DH=64 DF=4096
#define BB  4
#define SS  2048
#define DD  1024
#define HH  16
#define DHH 64
#define DFF 4096
#define MM  (BB * SS)   // 8192 rows

typedef __attribute__((ext_vector_type(16))) __bf16 v16bf;
typedef __attribute__((ext_vector_type(8)))  float  v8f;

__device__ __forceinline__ v8f wmma_bf16(v16bf a, v16bf b, v8f c) {
  return __builtin_amdgcn_wmma_f32_16x16x32_bf16(false, a, false, b, (short)0, c,
                                                 false, false);
}

// ---- fragment loaders (contiguous-k order, wave32) -------------------------
// A 16x32 bf16, row-major source: lane L holds row M=L%16; elements j ->
// K = (L/16)*16 + j (lanes 16-31 add +16).
__device__ __forceinline__ v16bf load_a(const __bf16* A, int lda, int row0, int k0) {
  int lane = threadIdx.x & 31;
  const __bf16* base =
      A + (size_t)(row0 + (lane & 15)) * lda + k0 + ((lane >> 4) << 4);
  v16bf a;
#pragma unroll
  for (int v = 0; v < 8; ++v) {
    int kk = 2 * v;   // 0,2,...,14
    a[2 * v]     = base[kk];
    a[2 * v + 1] = base[kk + 1];
  }
  return a;
}

// B 32x16 where B[k][n] = Bt[n*ldt + k] (N-major source, global or LDS).
// Lane L holds col N=L%16; elements j -> K = (L/16)*16 + j (32B contiguous).
__device__ __forceinline__ v16bf load_bt(const __bf16* Bt, int ldt, int k0, int col0) {
  int lane = threadIdx.x & 31;
  const __bf16* base =
      Bt + (size_t)(col0 + (lane & 15)) * ldt + k0 + ((lane >> 4) << 4);
  v16bf b;
#pragma unroll
  for (int j = 0; j < 16; ++j) b[j] = base[j];
  return b;
}

// ---- async memory->LDS copy primitives (CDNA5, ASYNCcnt) -------------------
__device__ __forceinline__ void async_b128(const __bf16* gsrc, const __bf16* ldst) {
  unsigned loff = (unsigned)(size_t)ldst;   // generic LDS ptr: low 32b = offset
  asm volatile("global_load_async_to_lds_b128 %0, %1, off"
               :: "v"(loff), "v"(gsrc) : "memory");
}
__device__ __forceinline__ void wait_async_le2() {
  asm volatile("s_wait_asynccnt 0x2" ::: "memory");
}
__device__ __forceinline__ void wait_async_le4() {
  asm volatile("s_wait_asynccnt 0x4" ::: "memory");
}
__device__ __forceinline__ void wait_async_0() {
  asm volatile("s_wait_asynccnt 0x0" ::: "memory");
}

// ---- staged, double-buffered K-loop, 2 M-tiles per wave --------------------
// acc[0..3]: rows [row0, row0+16)  x  cols [col0, col0+64)
// acc[4..7]: rows [row0+128, +16)  x  cols [col0, col0+64)
// lds: 2 * 64 * 64 bf16 (16KB). All 256 threads participate (barriers!).
// LDS tile layout: [col 0..63][k 0..63] row-major (128B per col = 8 chunks).
__device__ __forceinline__ void gemm_loop_staged(const __bf16* __restrict__ A,
                                                 const __bf16* __restrict__ Bt,
                                                 int K, int row0, int col0,
                                                 __bf16* lds, v8f acc[8]) {
  const int tid = threadIdx.x;
  auto issue = [&](int kk, int buf) {
#pragma unroll
    for (int i = 0; i < 2; ++i) {               // 512 chunks / 256 threads
      int c = tid + i * 256;
      int col = c >> 3, part = (c & 7) * 8;     // 8 bf16 = 16B per chunk
      async_b128(Bt + (size_t)(col0 + col) * K + kk + part,
                 lds + buf * 4096 + col * 64 + part);
    }
  };
  issue(0, 0);
  v16bf a0 = load_a(A, K, row0, 0);
  v16bf a1 = load_a(A, K, row0, 32);
  v16bf a2 = load_a(A, K, row0 + 128, 0);
  v16bf a3 = load_a(A, K, row0 + 128, 32);
  int buf = 0;
  for (int k = 0; k < K; k += 64) {
    bool more = (k + 64) < K;
    if (more) { issue(k + 64, buf ^ 1); wait_async_le2(); }
    else      { wait_async_0(); }
    __syncthreads();
    v16bf a0n = a0, a1n = a1, a2n = a2, a3n = a3;
    if (more) {
      a0n = load_a(A, K, row0, k + 64);
      a1n = load_a(A, K, row0, k + 96);
      a2n = load_a(A, K, row0 + 128, k + 64);
      a3n = load_a(A, K, row0 + 128, k + 96);
    }
    const __bf16* tile = lds + buf * 4096;
#pragma unroll
    for (int n = 0; n < 4; ++n) {
      v16bf b0 = load_bt(tile, 64, 0,  n * 16);
      v16bf b1 = load_bt(tile, 64, 32, n * 16);
      acc[n]     = wmma_bf16(a0, b0, acc[n]);        // each B frag feeds 2 WMMAs
      acc[n + 4] = wmma_bf16(a2, b0, acc[n + 4]);
      acc[n]     = wmma_bf16(a1, b1, acc[n]);
      acc[n + 4] = wmma_bf16(a3, b1, acc[n + 4]);
    }
    __syncthreads();
    a0 = a0n; a1 = a1n; a2 = a2n; a3 = a3n; buf ^= 1;
  }
}

// ---- weight convert + transpose: src fp32 [Hn][K][N] -> dst bf16 [Hn][N][K]
__global__ __launch_bounds__(256) void cvt_t_kernel(const float* __restrict__ src,
                                                    __bf16* __restrict__ dst,
                                                    int Hn, int K, int N) {
  size_t total = (size_t)Hn * K * N;
  for (size_t i = (size_t)blockIdx.x * blockDim.x + threadIdx.x; i < total;
       i += (size_t)gridDim.x * blockDim.x) {
    size_t kn = (size_t)K * N;
    int h = (int)(i / kn);
    size_t rem = i - (size_t)h * kn;
    int k = (int)(rem / N);
    int n = (int)(rem - (size_t)k * N);
    dst[((size_t)h * N + n) * K + k] = (__bf16)src[i];
  }
}

// ---- LayerNorm row kernel: fp32 in -> bf16 out -----------------------------
__global__ __launch_bounds__(256) void ln_kernel(const float* __restrict__ x,
                                                 const float* __restrict__ w,
                                                 const float* __restrict__ bsh,
                                                 __bf16* __restrict__ out) {
  int row = blockIdx.x;
  const float* xr = x + (size_t)row * DD;
  float s = 0.f, s2 = 0.f;
  for (int i = threadIdx.x; i < DD; i += 256) {
    float v = xr[i];
    s += v;
    s2 += v * v;
  }
#pragma unroll
  for (int off = 16; off >= 1; off >>= 1) {
    s  += __shfl_xor(s, off, 32);
    s2 += __shfl_xor(s2, off, 32);
  }
  __shared__ float red[2][8];
  int wave = threadIdx.x >> 5;
  if ((threadIdx.x & 31) == 0) { red[0][wave] = s; red[1][wave] = s2; }
  __syncthreads();
  float ts = 0.f, ts2 = 0.f;
#pragma unroll
  for (int i = 0; i < 8; ++i) { ts += red[0][i]; ts2 += red[1][i]; }
  float mu   = ts * (1.f / DD);
  float var  = ts2 * (1.f / DD) - mu * mu;
  float rinv = rsqrtf(var + 1e-5f);
  for (int i = threadIdx.x; i < DD; i += 256)
    out[(size_t)row * DD + i] = (__bf16)((xr[i] - mu) * rinv * w[i] + bsh[i]);
}

// ---- QKV projection (staged): q/k [bh,S,DH], v^T [bh,DH,S] -----------------
__global__ __launch_bounds__(256) void qkv_kernel(
    const __bf16* __restrict__ hidden, const __bf16* __restrict__ wqt,
    const __bf16* __restrict__ wkt, const __bf16* __restrict__ wvt,
    const float* __restrict__ bq, const float* __restrict__ bk,
    const float* __restrict__ bv, __bf16* __restrict__ qo,
    __bf16* __restrict__ ko, __bf16* __restrict__ vto) {
  __shared__ __bf16 lds[2 * 4096];
  int mat  = blockIdx.z;                     // 0=q 1=k 2=v
  int h    = blockIdx.y;
  int wave = threadIdx.x >> 5;
  int row0 = blockIdx.x * 256 + wave * 16;   // block covers 256 rows
  const __bf16* Wt   = (mat == 0 ? wqt : mat == 1 ? wkt : wvt) + (size_t)h * DHH * DD;
  const float*  bias = (mat == 0 ? bq : mat == 1 ? bk : bv) + h * DHH;
  v8f acc[8] = {};
  gemm_loop_staged(hidden, Wt, DD, row0, 0, lds, acc);
  int lane = threadIdx.x & 31, half = lane >> 4, colL = lane & 15;
#pragma unroll
  for (int rt = 0; rt < 2; ++rt) {
    int rowb = row0 + rt * 128;
    int b = rowb / SS, s0 = rowb % SS;
    size_t bh = (size_t)(b * HH + h);
#pragma unroll
    for (int n = 0; n < 4; ++n) {
      float bvv = bias[n * 16 + colL];
#pragma unroll
      for (int r = 0; r < 8; ++r) {
        int   srow = s0 + half * 8 + r;
        float v    = acc[rt * 4 + n][r] + bvv;
        if (mat == 0)
          qo[(bh * SS + srow) * DHH + n * 16 + colL] = (__bf16)v;
        else if (mat == 1)
          ko[(bh * SS + srow) * DHH + n * 16 + colL] = (__bf16)v;
        else
          vto[(bh * DHH + n * 16 + colL) * SS + srow] = (__bf16)v;  // V transposed
      }
    }
  }
}

// ---- Flash attention per (b,h): K/V tiles async-staged to LDS --------------
__global__ __launch_bounds__(256) void attn_kernel(
    const __bf16* __restrict__ q, const __bf16* __restrict__ k,
    const __bf16* __restrict__ vt, const int* __restrict__ mask,
    __bf16* __restrict__ attended) {
  __shared__ __bf16 kt[2][64 * 64];          // [buf][key][dh]    16KB
  __shared__ __bf16 vl[2][64 * 64];          // [buf][dh][key]    16KB
  __shared__ __bf16 pt[8][16 * 64];          // per-wave P tile   16KB
  int tid  = threadIdx.x;
  int wave = tid >> 5;
  int lane = tid & 31;
  int half = lane >> 4, colL = lane & 15;
  int bh = blockIdx.y;
  int b = bh / HH, hh = bh % HH;
  int q0 = blockIdx.x * 128 + wave * 16;
  const __bf16* qb = q + (size_t)bh * SS * DHH;
  const __bf16* kb = k + (size_t)bh * SS * DHH;
  const __bf16* vb = vt + (size_t)bh * DHH * SS;
  const int*    mb = mask + b * SS;

  auto issue_kv = [&](int t0, int buf) {
#pragma unroll
    for (int i = 0; i < 2; ++i) {            // K tile: 512 chunks
      int c = tid + i * 256;
      int row = c >> 3, part = (c & 7) * 8;
      async_b128(kb + (size_t)(t0 + row) * DHH + part, &kt[buf][row * 64 + part]);
    }
#pragma unroll
    for (int i = 0; i < 2; ++i) {            // V tile: 512 chunks
      int c = tid + i * 256;
      int row = c >> 3, part = (c & 7) * 8;
      async_b128(vb + (size_t)row * SS + t0 + part, &vl[buf][row * 64 + part]);
    }
  };

  v16bf qa0 = load_a(qb, DHH, q0, 0);
  v16bf qa1 = load_a(qb, DHH, q0, 32);

  v8f   o[4] = {};
  float m[8], l[8];
#pragma unroll
  for (int r = 0; r < 8; ++r) { m[r] = -1e30f; l[r] = 0.f; }
  const float scale = 0.125f;                 // 1/sqrt(64)

  issue_kv(0, 0);
  int buf = 0;
  for (int t0 = 0; t0 < SS; t0 += 64) {
    bool more = (t0 + 64) < SS;
    if (more) { issue_kv(t0 + 64, buf ^ 1); wait_async_le4(); }
    else      { wait_async_0(); }
    __syncthreads();
    const __bf16* ktile = kt[buf];
    const __bf16* vtile = vl[buf];
    // scores: S = Q K^T
    v8f sa[4] = {};
#pragma unroll
    for (int n = 0; n < 4; ++n) {
      sa[n] = wmma_bf16(qa0, load_bt(ktile, 64, 0,  n * 16), sa[n]);
      sa[n] = wmma_bf16(qa1, load_bt(ktile, 64, 32, n * 16), sa[n]);
    }
    // mask + scale
    float p[4][8];
#pragma unroll
    for (int n = 0; n < 4; ++n) {
      int  t     = t0 + n * 16 + colL;
      bool valid = (mb[t] != 0);
#pragma unroll
      for (int r = 0; r < 8; ++r)
        p[n][r] = valid ? sa[n][r] * scale : -1e9f;
    }
    // online softmax (row stats live in the 16-lane half: xor 1,2,4,8)
    float newm[8];
#pragma unroll
    for (int r = 0; r < 8; ++r) {
      float mx = fmaxf(fmaxf(p[0][r], p[1][r]), fmaxf(p[2][r], p[3][r]));
#pragma unroll
      for (int off = 1; off < 16; off <<= 1) mx = fmaxf(mx, __shfl_xor(mx, off, 32));
      newm[r] = fmaxf(m[r], mx);
    }
#pragma unroll
    for (int r = 0; r < 8; ++r) {
      float alpha = __expf(m[r] - newm[r]);
      l[r] *= alpha;
#pragma unroll
      for (int d = 0; d < 4; ++d) o[d][r] *= alpha;
      m[r] = newm[r];
      float sum = 0.f;
#pragma unroll
      for (int n = 0; n < 4; ++n) {
        p[n][r] = __expf(p[n][r] - newm[r]);
        sum += p[n][r];
      }
#pragma unroll
      for (int off = 1; off < 16; off <<= 1) sum += __shfl_xor(sum, off, 32);
      l[r] += sum;
    }
    // stage P -> per-wave LDS (row-major 16x64), reload as A fragments
    __bf16* pl = pt[wave];
#pragma unroll
    for (int n = 0; n < 4; ++n)
#pragma unroll
      for (int r = 0; r < 8; ++r)
        pl[(half * 8 + r) * 64 + n * 16 + colL] = (__bf16)p[n][r];
    asm volatile("s_wait_dscnt 0x0" ::: "memory");   // same-wave DS RAW fence
    v16bf pa0 = load_a(pl, 64, 0, 0);
    v16bf pa1 = load_a(pl, 64, 0, 32);
    // O += P V
#pragma unroll
    for (int d = 0; d < 4; ++d) {
      o[d] = wmma_bf16(pa0, load_bt(vtile, 64, 0,  d * 16), o[d]);
      o[d] = wmma_bf16(pa1, load_bt(vtile, 64, 32, d * 16), o[d]);
    }
    __syncthreads();
    buf ^= 1;
  }
  // normalize rows, store concat-head layout [B*S, D] bf16
#pragma unroll
  for (int r = 0; r < 8; ++r) {
    float inv = 1.f / l[r];
    int   srow = q0 + half * 8 + r;
#pragma unroll
    for (int d = 0; d < 4; ++d)
      attended[((size_t)b * SS + srow) * DD + hh * DHH + d * 16 + colL] =
          (__bf16)(o[d][r] * inv);
  }
}

// ---- generic staged WMMA GEMM with fused epilogue --------------------------
// MODE 0: out_f32 = A@B + bias + res      (Wo proj, FFN2)
// MODE 1: out_bf16 = gelu(A@B + bias)     (FFN1)
template <int MODE>
__global__ __launch_bounds__(256) void gemm_kernel(
    const __bf16* __restrict__ A, const __bf16* __restrict__ Bt,
    const float* __restrict__ bias, const float* __restrict__ res,
    float* __restrict__ outf, __bf16* __restrict__ outb, int N, int K) {
  __shared__ __bf16 lds[2 * 4096];
  int wave = threadIdx.x >> 5;
  int row0 = blockIdx.x * 256 + wave * 16;   // block covers 256 rows
  int col0 = blockIdx.y * 64;
  v8f acc[8] = {};
  gemm_loop_staged(A, Bt, K, row0, col0, lds, acc);
  int lane = threadIdx.x & 31, half = lane >> 4, colL = lane & 15;
#pragma unroll
  for (int rt = 0; rt < 2; ++rt) {
    int rowb = row0 + rt * 128;
#pragma unroll
    for (int n = 0; n < 4; ++n) {
      int   col = col0 + n * 16 + colL;
      float bvv = bias[col];
#pragma unroll
      for (int r = 0; r < 8; ++r) {
        size_t idx = (size_t)(rowb + half * 8 + r) * N + col;
        float  v   = acc[rt * 4 + n][r] + bvv;
        if (MODE == 0) {
          outf[idx] = v + res[idx];
        } else {
          v = 0.5f * v * (1.f + erff(v * 0.70710678118654752f));   // exact gelu
          outb[idx] = (__bf16)v;
        }
      }
    }
  }
}

// ---- host side -------------------------------------------------------------
extern "C" void kernel_launch(void* const* d_in, const int* in_sizes, int n_in,
                              void* d_out, int out_size, void* d_ws,
                              size_t ws_size, hipStream_t stream) {
  const float* state = (const float*)d_in[0];
  const int*   mask  = (const int*)d_in[1];
  const float* ln1w  = (const float*)d_in[2];
  const float* ln1b  = (const float*)d_in[3];
  const float* Wq    = (const float*)d_in[4];
  const float* bq    = (const float*)d_in[5];
  const float* Wk    = (const float*)d_in[6];
  const float* bk    = (const float*)d_in[7];
  const float* Wv    = (const float*)d_in[8];
  const float* bv    = (const float*)d_in[9];
  const float* Wo    = (const float*)d_in[10];
  const float* bo    = (const float*)d_in[11];
  const float* ln2w  = (const float*)d_in[12];
  const float* ln2b  = (const float*)d_in[13];
  const float* W1    = (const float*)d_in[14];
  const float* b1    = (const float*)d_in[15];
  const float* W2    = (const float*)d_in[16];
  const float* b2    = (const float*)d_in[17];
  (void)in_sizes; (void)n_in; (void)out_size; (void)ws_size;

  char* wsp = (char*)d_ws;
  auto alloc = [&](size_t bytes) -> char* {
    char* p = wsp;
    wsp += (bytes + 255) & ~(size_t)255;
    return p;
  };
  __bf16* hidden  = (__bf16*)alloc((size_t)MM * DD * 2);
  __bf16* wqt     = (__bf16*)alloc((size_t)HH * DD * DHH * 2);
  __bf16* wkt     = (__bf16*)alloc((size_t)HH * DD * DHH * 2);
  __bf16* wvt     = (__bf16*)alloc((size_t)HH * DD * DHH * 2);
  __bf16* wot     = (__bf16*)alloc((size_t)DD * DD * 2);
  __bf16* w1t     = (__bf16*)alloc((size_t)DD * DFF * 2);
  __bf16* w2t     = (__bf16*)alloc((size_t)DFF * DD * 2);
  __bf16* qbuf    = (__bf16*)alloc((size_t)BB * HH * SS * DHH * 2);
  __bf16* kbuf    = (__bf16*)alloc((size_t)BB * HH * SS * DHH * 2);
  __bf16* vtbuf   = (__bf16*)alloc((size_t)BB * HH * DHH * SS * 2);
  __bf16* attend  = (__bf16*)alloc((size_t)MM * DD * 2);
  float*  output  = (float*)alloc((size_t)MM * DD * 4);
  __bf16* hidden2 = (__bf16*)alloc((size_t)MM * DD * 2);
  __bf16* ffbuf   = (__bf16*)alloc((size_t)MM * DFF * 2);

  // 1. weight convert+transpose to bf16 N-major
  cvt_t_kernel<<<4096, 256, 0, stream>>>(Wq, wqt, HH, DD, DHH);
  cvt_t_kernel<<<4096, 256, 0, stream>>>(Wk, wkt, HH, DD, DHH);
  cvt_t_kernel<<<4096, 256, 0, stream>>>(Wv, wvt, HH, DD, DHH);
  cvt_t_kernel<<<4096, 256, 0, stream>>>(Wo, wot, 1, DD, DD);
  cvt_t_kernel<<<8192, 256, 0, stream>>>(W1, w1t, 1, DD, DFF);
  cvt_t_kernel<<<8192, 256, 0, stream>>>(W2, w2t, 1, DFF, DD);
  // 2. LN1 -> bf16 hidden
  ln_kernel<<<MM, 256, 0, stream>>>(state, ln1w, ln1b, hidden);
  // 3. QKV projections (WMMA, async-staged B, 2 M-tiles/wave)
  qkv_kernel<<<dim3(MM / 256, HH, 3), 256, 0, stream>>>(
      hidden, wqt, wkt, wvt, bq, bk, bv, qbuf, kbuf, vtbuf);
  // 4. flash attention (async-staged K/V, WMMA scores + WMMA PV)
  attn_kernel<<<dim3(SS / 128, BB * HH), 256, 0, stream>>>(
      qbuf, kbuf, vtbuf, mask, attend);
  // 5. output proj + bias + residual -> fp32 "output"
  gemm_kernel<0><<<dim3(MM / 256, DD / 64), 256, 0, stream>>>(
      attend, wot, bo, state, output, nullptr, DD, DD);
  // 6. LN2 -> bf16 hidden2
  ln_kernel<<<MM, 256, 0, stream>>>(output, ln2w, ln2b, hidden2);
  // 7. FFN1 + gelu -> bf16
  gemm_kernel<1><<<dim3(MM / 256, DFF / 64), 256, 0, stream>>>(
      hidden2, w1t, b1, nullptr, nullptr, ffbuf, DFF, DD);
  // 8. FFN2 + bias + residual -> d_out fp32
  gemm_kernel<0><<<dim3(MM / 256, DD / 64), 256, 0, stream>>>(
      ffbuf, w2t, b2, output, (float*)d_out, nullptr, DD, DFF);
}
